// StageNet_9199819948523
// MI455X (gfx1250) — matrix operand order repaired and checked by reading, not compiled, not run on hardware
//
#include <hip/hip_runtime.h>
#include <hip/hip_bf16.h>
#include <math.h>

typedef __attribute__((ext_vector_type(16))) _Float16 v16h;
typedef __attribute__((ext_vector_type(8)))  float    v8f;

#define B_  2
#define V_  5
#define C_  32
#define H_  256
#define W_  320
#define D_  8
#define HW_ (H_*W_)

// 4-byte-aligned float pair for single-request bilinear x-tap loads.
struct __align__(4) F2 { float a, b; };

// fast hardware reciprocal / exp (1-ulp class, plenty for sampling coords
// and softmax weights; avoids the ~10-instr IEEE div / expf fixup chains)
__device__ __forceinline__ float fastrcp(float x) { return __builtin_amdgcn_rcpf(x); }
__device__ __forceinline__ float fastexp(float x) {
    return __builtin_amdgcn_exp2f(x * 1.4426950408889634f);
}

// ------------------------------------------------------------------
// Setup kernel: per (b, v=1..4) compute rot(3x3)+trans(3) of
// src_proj @ inv(ref_proj), where proj = fuse(intr, extr).
// ws layout: [b][v-1][12] floats.
// ------------------------------------------------------------------
__global__ void proj_setup_kernel(const float* __restrict__ pm,
                                  float* __restrict__ ws)
{
    int b = threadIdx.x;
    if (b >= B_) return;

    auto fuse = [&](int v, float M[4][4]) {
        const float* extr = pm + (((b*V_ + v)*2 + 0)*16);
        const float* intr = pm + (((b*V_ + v)*2 + 1)*16);
        for (int r = 0; r < 3; ++r)
            for (int c = 0; c < 4; ++c) {
                float s = 0.f;
                for (int k = 0; k < 3; ++k) s += intr[r*4+k]*extr[k*4+c];
                M[r][c] = s;
            }
        for (int c = 0; c < 4; ++c) M[3][c] = extr[3*4+c];
    };

    float ref[4][4]; fuse(0, ref);

    // 4x4 inverse via Gauss-Jordan w/ partial pivoting (exact IEEE div here:
    // runs once per batch, accuracy matters most in this matrix)
    float inv[4][4];
    {
        float a[4][8];
        for (int r=0;r<4;++r)
            for (int c=0;c<4;++c){ a[r][c]=ref[r][c]; a[r][c+4]=(r==c)?1.f:0.f; }
        for (int col=0; col<4; ++col) {
            int piv = col; float pv = fabsf(a[col][col]);
            for (int r=col+1;r<4;++r){ float q=fabsf(a[r][col]); if (q>pv){pv=q;piv=r;} }
            if (piv != col)
                for (int c=0;c<8;++c){ float t=a[col][c]; a[col][c]=a[piv][c]; a[piv][c]=t; }
            float dinv = 1.f / a[col][col];
            for (int c=0;c<8;++c) a[col][c]*=dinv;
            for (int r=0;r<4;++r) if (r!=col){
                float f=a[r][col];
                for(int c=0;c<8;++c) a[r][c]-=f*a[col][c];
            }
        }
        for (int r=0;r<4;++r) for(int c=0;c<4;++c) inv[r][c]=a[r][c+4];
    }

    for (int v = 1; v < V_; ++v) {
        float S[4][4]; fuse(v, S);
        float P[3][4];
        for (int r=0;r<3;++r)
            for (int c=0;c<4;++c){
                float s=0.f;
                for(int k=0;k<4;++k) s += S[r][k]*inv[k][c];
                P[r][c]=s;
            }
        float* o = ws + (b*(V_-1) + (v-1))*12;
        o[0]=P[0][0]; o[1]=P[0][1]; o[2]=P[0][2];
        o[3]=P[1][0]; o[4]=P[1][1]; o[5]=P[1][2];
        o[6]=P[2][0]; o[7]=P[2][1]; o[8]=P[2][2];
        o[9]=P[0][3]; o[10]=P[1][3]; o[11]=P[2][3];
    }
}

// ------------------------------------------------------------------
// Main fused kernel.
// Wave = 16 pixels x 2 channel-halves (lane = pixel + 16*half).
// One v_wmma_f32_16x16x32_f16 per (view, depth) computes BOTH scalar
// reductions the algorithm needs, via a constant-ish A matrix:
//   S row 8 : 0.125            -> gsum = (sum_g mean_c(W*R))/attn_temp
//   S row 9 : 0.25*w_reg[k>>2] -> gdot = sum_g mean_c(W*R) * w_reg[g]
// Rows 8/9 of D land in VGPR 0/1 of lanes 16-31 (D layout: VGPR r,
// lanes16-31 = row 8+r), so lanes 16-31 own the per-pixel epilogue.
// The contiguous B k order used here, same as A (VGPR j: lanes0-15 K=2j,2j+1; lanes16-31
// K=16+2j,16+2j+1) matches per-lane channel ownership -> no shuffles.
// ------------------------------------------------------------------
__global__ __launch_bounds__(256)
void stagenet_main_kernel(const float* __restrict__ features,
                          const float* __restrict__ depth_hypo,
                          const float* __restrict__ w_reg,
                          const float* __restrict__ projws,
                          const int*   __restrict__ ratio_p,
                          float* __restrict__ out)
{
    const int tid  = threadIdx.x;
    const int lane = tid & 31;
    const int wv   = tid >> 5;          // wave id in block (0..7)
    const int mrow = lane & 15;         // pixel slot == WMMA row id
    const int half = lane >> 4;         // channel half (0: c0-15, 1: c16-31)

    const int b     = (int)((blockIdx.x * 128u) / (unsigned)HW_);  // block-uniform
    const int local = (int)(blockIdx.x * 128u) + wv * 16 + mrow - b * HW_;
    const int y     = local / W_;
    const int x     = local - y * W_;

    // A matrix: row 8 = 0.125 (group-mean 1/4 * attn-temp 1/2 folded),
    //           row 9 = 0.25 * w_reg[group], all other rows 0.
    // A k order used here (same as B, not AMD's table order): VGPR j: K=2j,2j+1 (lanes0-15) /
    //                  K=16+2j,16+2j+1 (lanes16-31).
    v16h amat;
    #pragma unroll
    for (int j = 0; j < 8; ++j) {
        const int k0 = 2*j + (half ? 16 : 0);
        float av = (mrow == 8) ? 0.125f
                 : (mrow == 9) ? 0.25f * w_reg[k0 >> 2]
                 : 0.0f;                       // k0,k0+1 share a group (k0 even)
        amat[2*j+0] = (_Float16)av;
        amat[2*j+1] = (_Float16)av;
    }

    // Reference features for this lane's 16 channels
    float R[16];
    const float* refbase = features
        + ((size_t)(b*V_ + 0)*C_ + half*16)*(size_t)HW_ + (size_t)y*W_ + x;
    #pragma unroll
    for (int i = 0; i < 16; ++i) R[i] = refbase[(size_t)i*HW_];

    // Depth hypotheses for this pixel
    float dh[D_];
    #pragma unroll
    for (int d = 0; d < D_; ++d)
        dh[d] = depth_hypo[((size_t)(b*D_ + d))*HW_ + (size_t)y*W_ + x];

    // cost_reg is linear in cor_feats: only need per-depth accumulators
    float acc[D_], wsum[D_];
    #pragma unroll
    for (int d = 0; d < D_; ++d) { acc[d] = 0.f; wsum[d] = 1e-8f; }

    const float fx = (float)x, fy = (float)y;

    #pragma unroll 1
    for (int v = 1; v < V_; ++v) {
        const float* pr = projws + (b*(V_-1) + (v-1))*12;
        const float r00=pr[0], r01=pr[1], r02=pr[2];
        const float r10=pr[3], r11=pr[4], r12=pr[5];
        const float r20=pr[6], r21=pr[7], r22=pr[8];
        const float t0 =pr[9], t1 =pr[10], t2 =pr[11];
        const float bx = r00*fx + r01*fy + r02;
        const float by = r10*fx + r11*fy + r12;
        const float bz = r20*fx + r21*fy + r22;
        const float* src = features + ((size_t)(b*V_ + v)*C_ + half*16)*(size_t)HW_;

        float gsum[D_], gdot[D_];
        #pragma unroll
        for (int d = 0; d < D_; ++d) {
            const float dep = dh[d];
            const float pz  = bz*dep + t2;
            const float iz  = fastrcp(pz);          // v_rcp_f32, 1-ulp
            const float px  = (bx*dep + t0) * iz;
            const float py  = (by*dep + t1) * iz;
            const float x0f = floorf(px), y0f = floorf(py);
            const float wx  = px - x0f,  wyf = py - y0f;
            const float mx0 = (x0f >=  0.f && x0f <= (float)(W_-1)) ? 1.f : 0.f;
            const float mx1 = (x0f >= -1.f && x0f <= (float)(W_-2)) ? 1.f : 0.f;
            const float my0 = (y0f >=  0.f && y0f <= (float)(H_-1)) ? 1.f : 0.f;
            const float my1 = (y0f >= -1.f && y0f <= (float)(H_-2)) ? 1.f : 0.f;
            // clamped pair start for single 8B x-tap load per row
            const float xsf = fminf(fmaxf(x0f, 0.f), (float)(W_-2));
            const bool  selx = (x0f == xsf);           // x0 in [0, W-2]
            const int   xs  = (int)xsf;
            const int y0c = (int)fminf(fmaxf(y0f,     0.f), (float)(H_-1));
            const int y1c = (int)fminf(fmaxf(y0f+1.f, 0.f), (float)(H_-1));
            const float w00 = (1.f-wx)*(1.f-wyf)*mx0*my0;
            const float w01 = wx      *(1.f-wyf)*mx1*my0;
            const float w10 = (1.f-wx)*wyf      *mx0*my1;
            const float w11 = wx      *wyf      *mx1*my1;
            // swap weights (not values) when the pair is shifted by clamping
            const float wa0 = selx ? w00 : w01, wb0 = selx ? w01 : w00;
            const float wa1 = selx ? w10 : w11, wb1 = selx ? w11 : w10;
            const int o0 = y0c*W_ + xs, o1 = y1c*W_ + xs;

            // Build B matrix: lane's 16 warped*ref products, packed 2/VGPR
            v16h bmat;
            #pragma unroll
            for (int i = 0; i < 16; ++i) {
                const float* p = src + (size_t)i*HW_;
                const F2 q0 = *reinterpret_cast<const F2*>(p + o0);
                const F2 q1 = *reinterpret_cast<const F2*>(p + o1);
                const float wval = wa0*q0.a + wb0*q0.b + wa1*q1.a + wb1*q1.b;
                bmat[i] = (_Float16)(wval * R[i]);
            }

            v8f cz = {};
            v8f dres = __builtin_amdgcn_wmma_f32_16x16x32_f16(
                false, amat, false, bmat, (short)0, cz, false, false);

            // lanes16-31: dres[0] = row8 = gsum/2, dres[1] = row9 = gdot
            gsum[d] = dres[0];
            gdot[d] = dres[1];
        }

        // cor_weight = softmax_d(gsum) / sqrt(C)   (attn_temp already folded)
        float mmax = gsum[0];
        #pragma unroll
        for (int d = 1; d < D_; ++d) mmax = fmaxf(mmax, gsum[d]);
        float e[D_], esum = 0.f;
        #pragma unroll
        for (int d = 0; d < D_; ++d) { e[d] = fastexp(gsum[d]-mmax); esum += e[d]; }
        const float sc = 0.17677669529663687f * fastrcp(esum);   // 1/sqrt(32)
        #pragma unroll
        for (int d = 0; d < D_; ++d) {
            const float wgt = e[d]*sc;
            wsum[d] += wgt;
            acc[d]  += wgt * gdot[d];
        }
    }

    // Per-pixel epilogue on lanes 16-31 (rows 8/9 of D live there);
    // lanes 0-15 computed on zero rows -> finite junk, skipped here.
    if (lane >= 16) {
        float cost[D_];
        #pragma unroll
        for (int d = 0; d < D_; ++d) cost[d] = acc[d] * fastrcp(wsum[d]);
        float cm = cost[0];
        #pragma unroll
        for (int d = 1; d < D_; ++d) cm = fmaxf(cm, cost[d]);
        float pe[D_], es = 0.f;
        #pragma unroll
        for (int d = 0; d < D_; ++d) { pe[d] = fastexp(cost[d]-cm); es += pe[d]; }
        const float inv_es = fastrcp(es);

        // first-occurrence argmax; max prob == exp(0)*inv_es == inv_es
        float best = cost[0], bdepth = dh[0];
        #pragma unroll
        for (int d = 1; d < D_; ++d)
            if (cost[d] > best) { best = cost[d]; bdepth = dh[d]; }

        float* out_depth = out;
        float* out_conf  = out + (size_t)B_*HW_;
        float* out_prob  = out + (size_t)2*B_*HW_;
        float* out_imin  = out + (size_t)2*B_*HW_ + (size_t)B_*D_*HW_;
        float* out_imax  = out_imin + (size_t)B_*HW_;
        const size_t pix = (size_t)b*HW_ + (size_t)y*W_ + x;

        #pragma unroll
        for (int d = 0; d < D_; ++d)
            out_prob[((size_t)(b*D_ + d))*HW_ + (size_t)y*W_ + x] = pe[d]*inv_es;

        const float itv   = fastrcp(dh[2]) - fastrcp(dh[1]);
        const float ratio = (float)ratio_p[0];
        const float idep  = fastrcp(bdepth);
        out_depth[pix] = bdepth;
        out_conf[pix]  = inv_es;
        out_imin[pix]  = idep + ratio*itv;
        out_imax[pix]  = idep - ratio*itv;
    }
}

// ------------------------------------------------------------------
extern "C" void kernel_launch(void* const* d_in, const int* in_sizes, int n_in,
                              void* d_out, int out_size, void* d_ws, size_t ws_size,
                              hipStream_t stream)
{
    const float* features = (const float*)d_in[0];
    const float* pm       = (const float*)d_in[1];
    const float* dh       = (const float*)d_in[2];
    const float* wreg     = (const float*)d_in[3];
    const int*   ratio    = (const int*)  d_in[6];
    float* out = (float*)d_out;
    float* ws  = (float*)d_ws;   // 96 floats used

    proj_setup_kernel<<<1, 32, 0, stream>>>(pm, ws);
    // 163840 pixels / (8 waves * 16 pixels) = 1280 blocks of 256 threads
    stagenet_main_kernel<<<(B_*H_*W_)/128, 256, 0, stream>>>(
        features, dh, wreg, ws, ratio, out);
}
